// CRF_89515708383904
// MI455X (gfx1250) — matrix operand order repaired and checked by reading, not compiled, not run on hardware
//
#include <hip/hip_runtime.h>
#include <hip/hip_bf16.h>

#define B_ 128
#define S_ 1024
#define L_ 256
#define NEG_INF (-3.0e38f)
#define LOG2E_F 1.4426950408889634f
#define LN2_F   0.6931471805599453f

typedef __attribute__((ext_vector_type(16))) __bf16 v16bf;
typedef __attribute__((ext_vector_type(8)))  __bf16 v8bf;
typedef __attribute__((ext_vector_type(8)))  float  v8f;
typedef unsigned int uint32x4 __attribute__((ext_vector_type(4)));
typedef int          int32x8  __attribute__((ext_vector_type(8)));
typedef int          int32x4  __attribute__((ext_vector_type(4)));

#if defined(__has_builtin)
#if __has_builtin(__builtin_amdgcn_tensor_load_to_lds) && __has_builtin(__builtin_amdgcn_s_wait_tensorcnt)
#define HAS_TDM 1
#endif
#endif

__device__ static inline unsigned short f2bf(float f) {
    __bf16 h = (__bf16)f;
    unsigned short u;
    __builtin_memcpy(&u, &h, 2);
    return u;
}

#ifdef HAS_TDM
// TDM: load a 16-row x 256-col f32 tile (row stride S*L elems) into LDS with
// 4-dword padding after every 256 dwords (-> 260-float LDS rows).
__device__ static inline void tdm_load_tile(const float* gsrc, unsigned lds_byte) {
    unsigned long long ga = (unsigned long long)(uintptr_t)gsrc;
    uint32x4 g0;
    g0[0] = 1u;                                              // count=1, no gather
    g0[1] = lds_byte;                                        // lds_addr
    g0[2] = (unsigned)(ga & 0xFFFFFFFFu);                    // global_addr lo
    g0[3] = (unsigned)((ga >> 32) & 0x01FFFFFFu) | (2u << 30); // addr hi | type=2
    int32x8 g1;
    g1[0] = (2 << 16)     // data_size = 4B
          | (1 << 20)     // pad_enable
          | (7 << 22)     // pad_interval: every 256 dwords
          | (3 << 25);    // pad_amount: 4 dwords
    g1[1] = (int)(256u << 16);   // tensor_dim0 = 256
    g1[2] = (int)(16u << 16);    // tensor_dim1 = 16
    g1[3] = (int)(256u << 16);   // tile_dim0 = 256
    g1[4] = 16;                  // tile_dim1 = 16, tile_dim2 = 0
    g1[5] = (int)(S_ * L_);      // tensor_dim0_stride = 262144 elems
    g1[6] = 0;
    g1[7] = 0;
    int32x4 z4 = {0, 0, 0, 0};
    int32x8 z8 = {0, 0, 0, 0, 0, 0, 0, 0};
    __builtin_amdgcn_tensor_load_to_lds(g0, g1, z4, z4, z8, 0);
}
#endif

// ---------------- gold (numerator) score: one block per batch ----------------
__global__ void crf_scores_kernel(const float* __restrict__ emissions,
                                  const int*   __restrict__ tags,
                                  const float* __restrict__ mask,
                                  const float* __restrict__ trans,
                                  float*       __restrict__ scores)
{
    const int b   = blockIdx.x;
    const int tid = threadIdx.x;
    __shared__ float sred[256];
    __shared__ int   ired[256];

    float part = 0.f;
    int   cnt  = 0;
    for (int t = tid; t < S_; t += 256) {
        float m  = mask[b * S_ + t];
        int   on = (m > 0.5f) ? 1 : 0;
        cnt += on;
        if (t >= 1 && on) {
            int tg = tags[b * S_ + t];
            int tp = tags[b * S_ + t - 1];
            part += emissions[(size_t)(b * S_ + t) * L_ + tg] + trans[tp * L_ + tg];
        }
    }
    sred[tid] = part;
    ired[tid] = cnt;
    __syncthreads();
    for (int s = 128; s > 0; s >>= 1) {
        if (tid < s) { sred[tid] += sred[tid + s]; ired[tid] += ired[tid + s]; }
        __syncthreads();
    }
    if (tid == 0) {
        int   last = ired[0] - 1;
        int   t0   = tags[b * S_];
        float sc   = trans[0 * L_ + t0]
                   + emissions[(size_t)(b * S_) * L_ + t0]
                   + sred[0];
        int lt = tags[b * S_ + last];
        sc += trans[lt * L_ + 1];
        scores[b] = sc;
    }
}

// ---------------- log-partition: forward scan as exp-domain WMMA GEMM --------
// grid = 8 blocks (16 batches each), 512 threads = 16 waves, wave w owns N-tile w.
__launch_bounds__(512, 1)
__global__ void crf_partition_kernel(const float* __restrict__ emissions,
                                     const float* __restrict__ mask,
                                     const float* __restrict__ trans,
                                     const float* __restrict__ scores,
                                     float*       __restrict__ out)
{
    const int tid  = threadIdx.x;
    const int wave = tid >> 5;
    const int lane = tid & 31;
    const int b0   = blockIdx.x * 16;

    __shared__ __align__(16) float  alphaS[16][260];
    __shared__ __align__(16) float  eS[2][16][260];    // double-buffered e_t
    __shared__ __align__(16) __bf16 pS[16][264];       // exp(alpha - rowmax)
    __shared__ float TendS[256];
    __shared__ float mrowS[16];
    __shared__ float maskS[16];

    // ---- stationary B fragments: expT[K][N], bf16 32x16 B layout ----
    const int ncol = (wave << 4) + (lane & 15);
    const int koff = (lane < 16) ? 0 : 8;
    v16bf Bfrag[8];
#pragma unroll
    for (int kt = 0; kt < 8; ++kt) {
#pragma unroll
        for (int e = 0; e < 16; ++e) {
            int K = kt * 32 + koff + (e < 8 ? e : e + 8);
            Bfrag[kt][e] = (__bf16)__builtin_amdgcn_exp2f(trans[K * L_ + ncol] * LOG2E_F);
        }
    }

    if (tid < 256) TendS[tid] = trans[tid * L_ + 1];   // T[:, EOS]

    // ---- alpha0 = T[BOS,:] + emissions[:,0,:] ----
    for (int idx = tid; idx < 16 * L_; idx += 512) {
        int r = idx >> 8, c = idx & 255;
        alphaS[r][c] = trans[c] + emissions[((size_t)(b0 + r) * S_) * L_ + c];
    }

    // ---- prologue prefetch of e_{t=1} into buffer 1 ----
#ifdef HAS_TDM
    if (wave == 0)
        tdm_load_tile(emissions + ((size_t)b0 * S_ + 1) * L_,
                      (unsigned)(uintptr_t)(&eS[1][0][0]));
#else
    float ebuf[8];
#pragma unroll
    for (int j = 0; j < 8; ++j) {
        int idx = tid + j * 512;
        int r = idx >> 8, c = idx & 255;
        ebuf[j] = emissions[((size_t)(b0 + r) * S_ + 1) * L_ + c];
    }
#pragma unroll
    for (int j = 0; j < 8; ++j) {
        int idx = tid + j * 512;
        eS[1][idx >> 8][idx & 255] = ebuf[j];
    }
#endif
    if (tid < 16) maskS[tid] = mask[(b0 + tid) * S_ + 1];
    __syncthreads();

    // renorm: per-row max via wave butterfly, P = exp(alpha - max) packed bf16
    auto renorm = [&]() {
        const int cb = 2 * lane;
        float av[8];
#pragma unroll
        for (int k = 0; k < 4; ++k) {
            av[2 * k]     = alphaS[wave][cb + 64 * k];
            av[2 * k + 1] = alphaS[wave][cb + 64 * k + 1];
        }
        float mx = av[0];
#pragma unroll
        for (int j = 1; j < 8; ++j) mx = fmaxf(mx, av[j]);
#pragma unroll
        for (int off = 16; off > 0; off >>= 1)
            mx = fmaxf(mx, __shfl_xor(mx, off, 32));
        if (lane == 0) mrowS[wave] = mx;
#pragma unroll
        for (int k = 0; k < 4; ++k) {
            float p0 = __builtin_amdgcn_exp2f((av[2 * k]     - mx) * LOG2E_F);
            float p1 = __builtin_amdgcn_exp2f((av[2 * k + 1] - mx) * LOG2E_F);
            unsigned u = (unsigned)f2bf(p0) | ((unsigned)f2bf(p1) << 16);
            *(unsigned*)&pS[wave][cb + 64 * k] = u;
        }
    };

    renorm();
#ifdef HAS_TDM
    if (wave == 0) __builtin_amdgcn_s_wait_tensorcnt(0);
#endif
    __syncthreads();   // publish pS, mrowS, maskS, eS[1]

    const int abase  = (lane < 16) ? 0 : 8;
    const int mbaseM = (lane < 16) ? 0 : 8;

    for (int t = 1; t < S_; ++t) {
        const int cur = t & 1;
        const int nxt = cur ^ 1;

        // kick off DMA of e_{t+1} into the other buffer; overlaps the GEMM
#ifdef HAS_TDM
        if (wave == 0 && t + 1 < S_)
            tdm_load_tile(emissions + ((size_t)b0 * S_ + (t + 1)) * L_,
                          (unsigned)(uintptr_t)(&eS[nxt][0][0]));
#else
        float ebuf[8];
        if (t + 1 < S_) {
#pragma unroll
            for (int j = 0; j < 8; ++j) {
                int idx = tid + j * 512;
                int r = idx >> 8, c = idx & 255;
                ebuf[j] = emissions[((size_t)(b0 + r) * S_ + (t + 1)) * L_ + c];
            }
        }
#endif

        // s = P @ expT for this wave's 16 columns, K=256 in 8 WMMA
        v8f acc = {0.f, 0.f, 0.f, 0.f, 0.f, 0.f, 0.f, 0.f};
        const __bf16* prow = &pS[lane & 15][0];
#pragma unroll
        for (int kt = 0; kt < 8; ++kt) {
            const __bf16* rp = prow + kt * 32 + abase;
            v8bf lo = *(const v8bf*)(rp);
            v8bf hi = *(const v8bf*)(rp + 16);
            v16bf A = __builtin_shufflevector(lo, hi, 0, 1, 2, 3, 4, 5, 6, 7,
                                              8, 9, 10, 11, 12, 13, 14, 15);
            acc = __builtin_amdgcn_wmma_f32_16x16x32_bf16(
                false, A, false, Bfrag[kt], (short)0, acc, false, false);
        }

        // alpha' = alpha + mask*(rowmax + log s + e_t - alpha)
#pragma unroll
        for (int v = 0; v < 8; ++v) {
            int   M  = v + mbaseM;
            float na = mrowS[M] + __builtin_amdgcn_logf(acc[v]) * LN2_F
                     + eS[cur][M][ncol];
            float mk = maskS[M];
            float oa = alphaS[M][ncol];
            alphaS[M][ncol] = oa + mk * (na - oa);
        }
        __syncthreads();   // alphaS complete

        renorm();
        if (tid < 16 && t + 1 < S_)
            maskS[tid] = mask[(b0 + tid) * S_ + (t + 1)];
#ifdef HAS_TDM
        if (wave == 0 && t + 1 < S_) __builtin_amdgcn_s_wait_tensorcnt(0);
#else
        if (t + 1 < S_) {
#pragma unroll
            for (int j = 0; j < 8; ++j) {
                int idx = tid + j * 512;
                eS[nxt][idx >> 8][idx & 255] = ebuf[j];
            }
        }
#endif
        __syncthreads();   // publish pS, mrowS, maskS, eS[nxt]
    }

    // ---- logZ = logsumexp_j(alpha + T[j,EOS]); out = logZ - gold score ----
    {
        const int cb = 2 * lane;
        float av[8];
#pragma unroll
        for (int k = 0; k < 4; ++k) {
            int c = cb + 64 * k;
            av[2 * k]     = alphaS[wave][c]     + TendS[c];
            av[2 * k + 1] = alphaS[wave][c + 1] + TendS[c + 1];
        }
        float mx = av[0];
#pragma unroll
        for (int j = 1; j < 8; ++j) mx = fmaxf(mx, av[j]);
#pragma unroll
        for (int off = 16; off > 0; off >>= 1)
            mx = fmaxf(mx, __shfl_xor(mx, off, 32));
        float sm = 0.f;
#pragma unroll
        for (int j = 0; j < 8; ++j)
            sm += __builtin_amdgcn_exp2f((av[j] - mx) * LOG2E_F);
#pragma unroll
        for (int off = 16; off > 0; off >>= 1)
            sm += __shfl_xor(sm, off, 32);
        if (lane == 0)
            out[b0 + wave] = mx + __builtin_amdgcn_logf(sm) * LN2_F
                           - scores[b0 + wave];
    }
}

extern "C" void kernel_launch(void* const* d_in, const int* in_sizes, int n_in,
                              void* d_out, int out_size, void* d_ws, size_t ws_size,
                              hipStream_t stream) {
    const float* emissions = (const float*)d_in[0];
    const int*   tags      = (const int*)  d_in[1];
    const float* mask      = (const float*)d_in[2];
    const float* trans     = (const float*)d_in[3];
    float*       out       = (float*)d_out;
    float*       scores    = (float*)d_ws;

    crf_scores_kernel<<<dim3(B_), dim3(256), 0, stream>>>(emissions, tags, mask, trans, scores);
    crf_partition_kernel<<<dim3(B_ / 16), dim3(512), 0, stream>>>(emissions, mask, trans, scores, out);
}
